// DRA_C_65644280152592
// MI455X (gfx1250) — matrix operand order repaired and checked by reading, not compiled, not run on hardware
//
#include <hip/hip_runtime.h>

typedef __attribute__((ext_vector_type(16))) __bf16 bf16x16;
typedef __attribute__((ext_vector_type(8)))  __bf16 bf16x8;
typedef __attribute__((ext_vector_type(8)))  float  f32x8;

#define LSTR 40   // LDS row stride in bf16 elements (64B row + 16B pad)

// ---- TDM (Tensor Data Mover) support, probe-guarded ----
#if defined(__has_builtin)
#if __has_builtin(__builtin_amdgcn_tensor_load_to_lds) && __has_builtin(__builtin_amdgcn_s_wait_tensorcnt)
#define ENABLE_TDM 1
#endif
#endif
#ifndef ENABLE_TDM
#define ENABLE_TDM 0
#endif

#if ENABLE_TDM
typedef __attribute__((ext_vector_type(4))) unsigned int u32x4;
typedef __attribute__((ext_vector_type(8))) int i32x8;
typedef __attribute__((ext_vector_type(4))) int i32x4;

// DMA a (rows x 32) bf16 tile (row stride = `stride` elements) from global into
// LDS at byte offset lds_off, writing 64B rows padded by 16B (-> 80B = LSTR*2).
// OOB rows/cols (beyond tensor dims) are zero-filled by the TDM.
static __device__ __forceinline__ void tdm_load_a_tile(
    const __bf16* gsrc, unsigned lds_off, int dim0 /*valid cols from tile*/,
    int dim1 /*valid rows from tile*/, int stride /*lda elements*/)
{
  unsigned long ga = (unsigned long)gsrc;
  u32x4 g0;
  g0[0] = 1u;                                            // count=1, user D#
  g0[1] = lds_off;                                       // LDS byte address
  g0[2] = (unsigned)(ga & 0xffffffffu);                  // global addr [31:0]
  g0[3] = (unsigned)((ga >> 32) & 0x1ffffffu) | (2u << 30); // addr[56:32] | type=2
  i32x8 g1;
  // data_size=1 (2B) | pad_enable | pad_interval=3 (64B) | pad_amount=3 (16B)
  g1[0] = (1 << 16) | (1 << 20) | (3 << 22) | (3 << 25);
  g1[1] = (int)(((unsigned)dim0 & 0xffffu) << 16);       // tensor_dim0[15:0] -> bits 63:48
  g1[2] = (int)((((unsigned)dim0 >> 16) & 0xffffu) |
                (((unsigned)dim1 & 0xffffu) << 16));     // dim0[31:16] | dim1[15:0]
  g1[3] = (int)((((unsigned)dim1 >> 16) & 0xffffu) |
                (32u << 16));                            // dim1[31:16] | tile_dim0=32
  g1[4] = 128;                                           // tile_dim1=128, tile_dim2=0
  g1[5] = stride;                                        // tensor_dim0_stride[31:0]
  g1[6] = 0;                                             // stride[47:32] | dim1_stride lo
  g1[7] = 0;
  i32x4 z4 = {0, 0, 0, 0};
#if __clang_major__ >= 23
  i32x8 z8 = {0, 0, 0, 0, 0, 0, 0, 0};
  __builtin_amdgcn_tensor_load_to_lds(g0, g1, z4, z4, z8, 0);
#else
  __builtin_amdgcn_tensor_load_to_lds(g0, g1, z4, z4, 0);
#endif
}
#endif

static __device__ __forceinline__ f32x8 wmma_bf16(bf16x16 a, bf16x16 b, f32x8 c) {
  return __builtin_amdgcn_wmma_f32_16x16x32_bf16(false, a, false, b, (short)0, c, false, false);
}
// A fragment: lane holds row (lane&15); ks = kb..kb+15 contiguous, kb=(lane<16)?0:16
static __device__ __forceinline__ void load_frag_a(bf16x16& f, const __bf16* p) {
  ((bf16x8*)&f)[0] = *(const bf16x8*)p;
  ((bf16x8*)&f)[1] = *(const bf16x8*)(p + 8);
}
// B fragment: lane holds col (lane&15); ks = kb..kb+15 contiguous (transposed [n][k] tile)
static __device__ __forceinline__ void load_frag_b(bf16x16& f, const __bf16* p) {
  ((bf16x8*)&f)[0] = *(const bf16x8*)p;
  ((bf16x8*)&f)[1] = *(const bf16x8*)(p + 8);
}

// ---------------------------------------------------------------------------
// Generic bf16 GEMM: C[M,N] = A[M,K] * B[K,N], 128x64 tile, BK=32, 256 thr.
// mode 0: store bf16;  mode 2: store f32;  mode 3: store f32 relu(acc*p0[m]+p1[m])
// ---------------------------------------------------------------------------
__global__ __launch_bounds__(256) void gemm_bf16_k(
    const __bf16* __restrict__ A, const __bf16* __restrict__ Bm, void* __restrict__ Cm,
    int M, int N, int K, int lda, int ldb, int ldc,
    long sA, long sB, long sC, int mode,
    const float* __restrict__ p0, const float* __restrict__ p1)
{
  __shared__ __align__(16) __bf16 As[128 * LSTR];
  __shared__ __align__(16) __bf16 Bs[64 * LSTR];
  const int bz = blockIdx.z;
  const __bf16* Ab = A + (size_t)bz * sA;
  const __bf16* Bb = Bm + (size_t)bz * sB;
  const int m0 = blockIdx.x * 128, n0 = blockIdx.y * 64;
  const int tid = threadIdx.x, lane = tid & 31, w = tid >> 5;
  const int wm = (w >> 1) * 32, wn = (w & 1) * 32;
  const int bk = tid >> 3, bng = (tid & 7) * 8;
  const int mrow = lane & 15, half = lane >> 4;
#if ENABLE_TDM
  const unsigned AsOff = (unsigned)(size_t)&As[0];
#else
  const int arow = tid >> 1, akc = (tid & 1) * 16;
#endif

  f32x8 acc[2][2] = {};

  for (int k0 = 0; k0 < K; k0 += 32) {
    // ---- stage A tile (128 x 32), row stride LSTR in LDS ----
#if ENABLE_TDM
    if (w == 0)
      tdm_load_a_tile(Ab + (size_t)m0 * lda + k0, AsOff, K - k0, M - m0, lda);
#else
    {
      bf16x8 a0 = {}, a1 = {};
      int gma = m0 + arow;
      if (gma < M) {
        const __bf16* ap = Ab + (size_t)gma * lda + (k0 + akc);
        a0 = *(const bf16x8*)ap;
        a1 = *(const bf16x8*)(ap + 8);
        if (k0 + 32 < K) __builtin_prefetch(ap + 32, 0, 3);
      }
      *(bf16x8*)&As[arow * LSTR + akc]     = a0;
      *(bf16x8*)&As[arow * LSTR + akc + 8] = a1;
    }
#endif
    // ---- stage B tile transposed into [n][k] ----
    {
      bf16x8 bv = {};
      int gn0 = n0 + bng;
      const __bf16* bp = Bb + (size_t)(k0 + bk) * ldb + gn0;
      if (gn0 + 8 <= N) bv = *(const bf16x8*)bp;
      else for (int i = 0; i < 8; ++i) if (gn0 + i < N) bv[i] = bp[i];
      if (k0 + 32 < K) __builtin_prefetch(bp + (size_t)32 * ldb, 0, 3);
      for (int i = 0; i < 8; ++i) Bs[(bng + i) * LSTR + bk] = bv[i];
    }
#if ENABLE_TDM
    __builtin_amdgcn_s_wait_tensorcnt(0);
#endif
    __syncthreads();

    bf16x16 af[2], bq[2];
    load_frag_a(af[0], &As[(wm + mrow)      * LSTR + half * 16]);
    load_frag_a(af[1], &As[(wm + 16 + mrow) * LSTR + half * 16]);
    load_frag_b(bq[0], &Bs[(wn + mrow)      * LSTR + half * 16]);
    load_frag_b(bq[1], &Bs[(wn + 16 + mrow) * LSTR + half * 16]);
    acc[0][0] = wmma_bf16(af[0], bq[0], acc[0][0]);
    acc[0][1] = wmma_bf16(af[0], bq[1], acc[0][1]);
    acc[1][0] = wmma_bf16(af[1], bq[0], acc[1][0]);
    acc[1][1] = wmma_bf16(af[1], bq[1], acc[1][1]);
    __syncthreads();
  }

  for (int i = 0; i < 2; ++i)
    for (int j = 0; j < 2; ++j) {
      int gn = n0 + wn + j * 16 + mrow;
      int mb = m0 + wm + i * 16 + (half << 3);
      if (gn < N) {
        for (int r = 0; r < 8; ++r) {
          int gm = mb + r;
          if (gm < M) {
            float v = acc[i][j][r];
            size_t cidx = (size_t)bz * sC + (size_t)gm * ldc + gn;
            if (mode == 0)      ((__bf16*)Cm)[cidx] = (__bf16)v;
            else if (mode == 2) ((float*)Cm)[cidx] = v;
            else                ((float*)Cm)[cidx] = fmaxf(fmaf(v, p0[gm], p1[gm]), 0.f);
          }
        }
      }
    }
}

// ---------------------------------------------------------------------------
// Patch-embed GEMM: dl[b][d][n] = pe_w[d, 32768] * patches(decoder)[32768, 196] + pe_b
// tile 128 x 32, BK=32.  Output bf16, n padded to 224 with zeros.
// ---------------------------------------------------------------------------
__global__ __launch_bounds__(256) void pe_gemm_k(
    const __bf16* __restrict__ Aw, const float* __restrict__ dec,
    const float* __restrict__ peb, __bf16* __restrict__ dl)
{
  __shared__ __align__(16) __bf16 As[128 * LSTR];
  __shared__ __align__(16) __bf16 Bs[32 * LSTR];
  const int b = blockIdx.z;
  const int m0 = blockIdx.x * 128, n0 = blockIdx.y * 32;
  const int tid = threadIdx.x, lane = tid & 31, w = tid >> 5;
  const int wm = (w >> 1) * 32, wn = (w & 1) * 16;
  const int bk = tid >> 3, bnq = (tid & 7) * 4;
  const int mrow = lane & 15, half = lane >> 4;
  const float* decb = dec + (size_t)b * 512 * 12544;
#if ENABLE_TDM
  const unsigned AsOff = (unsigned)(size_t)&As[0];
#else
  const int arow = tid >> 1, akc = (tid & 1) * 16;
#endif

  f32x8 acc[2] = {};

  for (int k0 = 0; k0 < 32768; k0 += 32) {
#if ENABLE_TDM
    if (w == 0)
      tdm_load_a_tile(Aw + (size_t)m0 * 32768 + k0, AsOff, 32768 - k0, 512 - m0, 32768);
#else
    {
      const __bf16* ap = Aw + (size_t)(m0 + arow) * 32768 + (k0 + akc);
      *(bf16x8*)&As[arow * LSTR + akc]     = *(const bf16x8*)ap;
      *(bf16x8*)&As[arow * LSTR + akc + 8] = *(const bf16x8*)(ap + 8);
      if (k0 + 32 < 32768) __builtin_prefetch(ap + 32, 0, 3);
    }
#endif
    int gk = k0 + bk;
    int ic = gk >> 6, rem = gk & 63, kh = rem >> 3, kw = rem & 7;
    const float* dp = decb + (size_t)ic * 12544 + kh * 112 + kw;
    for (int i = 0; i < 4; ++i) {
      int nl = bnq + i, gn = n0 + nl;
      __bf16 v = (__bf16)0.f;
      if (gn < 196) { int ph = gn / 14; int pw = gn - ph * 14; v = (__bf16)dp[ph * 896 + pw * 8]; }
      Bs[nl * LSTR + bk] = v;
    }
#if ENABLE_TDM
    __builtin_amdgcn_s_wait_tensorcnt(0);
#endif
    __syncthreads();

    bf16x16 af[2], bq;
    load_frag_a(af[0], &As[(wm + mrow)      * LSTR + half * 16]);
    load_frag_a(af[1], &As[(wm + 16 + mrow) * LSTR + half * 16]);
    load_frag_b(bq,    &Bs[(wn + mrow)      * LSTR + half * 16]);
    acc[0] = wmma_bf16(af[0], bq, acc[0]);
    acc[1] = wmma_bf16(af[1], bq, acc[1]);
    __syncthreads();
  }

  int gn = n0 + wn + mrow;
  for (int i = 0; i < 2; ++i) {
    int mb = m0 + wm + i * 16 + (half << 3);
    for (int r = 0; r < 8; ++r) {
      int gm = mb + r;
      float v = (gn < 196) ? (acc[i][r] + peb[gm]) : 0.f;
      dl[((size_t)b * 512 + gm) * 224 + gn] = (__bf16)v;
    }
  }
}

// ---------------------------------------------------------------------------
// Fused final: mask = relu(bn1(mc_w_eff @ decoder)); out = mask * z14[upsampled]
// GEMM M=512, K=512, N=12544/batch, tile 128x64.
// ---------------------------------------------------------------------------
__global__ __launch_bounds__(256) void mask_final_k(
    const __bf16* __restrict__ Aw, const float* __restrict__ dec,
    const float* __restrict__ b1e, const float* __restrict__ z14,
    float* __restrict__ out)
{
  __shared__ __align__(16) __bf16 As[128 * LSTR];
  __shared__ __align__(16) __bf16 Bs[64 * LSTR];
  const int b = blockIdx.z;
  const int m0 = blockIdx.x * 128, n0 = blockIdx.y * 64;
  const int tid = threadIdx.x, lane = tid & 31, w = tid >> 5;
  const int wm = (w >> 1) * 32, wn = (w & 1) * 32;
  const int bk = tid >> 3, bng = (tid & 7) * 8;
  const int mrow = lane & 15, half = lane >> 4;
  const float* decb = dec + (size_t)b * 512 * 12544;
#if ENABLE_TDM
  const unsigned AsOff = (unsigned)(size_t)&As[0];
#else
  const int arow = tid >> 1, akc = (tid & 1) * 16;
#endif

  f32x8 acc[2][2] = {};

  for (int k0 = 0; k0 < 512; k0 += 32) {
#if ENABLE_TDM
    if (w == 0)
      tdm_load_a_tile(Aw + (size_t)m0 * 512 + k0, AsOff, 512 - k0, 512 - m0, 512);
#else
    {
      const __bf16* ap = Aw + (size_t)(m0 + arow) * 512 + (k0 + akc);
      *(bf16x8*)&As[arow * LSTR + akc]     = *(const bf16x8*)ap;
      *(bf16x8*)&As[arow * LSTR + akc + 8] = *(const bf16x8*)(ap + 8);
    }
#endif
    const float* sp = decb + (size_t)(k0 + bk) * 12544 + n0 + bng;
    if (k0 + 32 < 512) __builtin_prefetch(sp + (size_t)32 * 12544, 0, 3);
    float4 f0 = *(const float4*)sp;
    float4 f1 = *(const float4*)(sp + 4);
    Bs[(bng + 0) * LSTR + bk] = (__bf16)f0.x;
    Bs[(bng + 1) * LSTR + bk] = (__bf16)f0.y;
    Bs[(bng + 2) * LSTR + bk] = (__bf16)f0.z;
    Bs[(bng + 3) * LSTR + bk] = (__bf16)f0.w;
    Bs[(bng + 4) * LSTR + bk] = (__bf16)f1.x;
    Bs[(bng + 5) * LSTR + bk] = (__bf16)f1.y;
    Bs[(bng + 6) * LSTR + bk] = (__bf16)f1.z;
    Bs[(bng + 7) * LSTR + bk] = (__bf16)f1.w;
#if ENABLE_TDM
    __builtin_amdgcn_s_wait_tensorcnt(0);
#endif
    __syncthreads();

    bf16x16 af[2], bq[2];
    load_frag_a(af[0], &As[(wm + mrow)      * LSTR + half * 16]);
    load_frag_a(af[1], &As[(wm + 16 + mrow) * LSTR + half * 16]);
    load_frag_b(bq[0], &Bs[(wn + mrow)      * LSTR + half * 16]);
    load_frag_b(bq[1], &Bs[(wn + 16 + mrow) * LSTR + half * 16]);
    acc[0][0] = wmma_bf16(af[0], bq[0], acc[0][0]);
    acc[0][1] = wmma_bf16(af[0], bq[1], acc[0][1]);
    acc[1][0] = wmma_bf16(af[1], bq[0], acc[1][0]);
    acc[1][1] = wmma_bf16(af[1], bq[1], acc[1][1]);
    __syncthreads();
  }

  for (int i = 0; i < 2; ++i)
    for (int j = 0; j < 2; ++j) {
      int p = n0 + wn + j * 16 + mrow;
      int h = p / 112, wp = p - h * 112;
      int np = (h >> 3) * 14 + (wp >> 3);
      int mb = m0 + wm + i * 16 + (half << 3);
      for (int r = 0; r < 8; ++r) {
        int gm = mb + r;
        float mask = fmaxf(acc[i][j][r] + b1e[gm], 0.f);
        out[((size_t)b * 512 + gm) * 12544 + p] = mask * z14[((size_t)b * 512 + gm) * 224 + np];
      }
    }
}

// ---------------------------------------------------------------------------
// Weight pack: f32 -> bf16 with optional transpose (strides), padding, row scale
// ---------------------------------------------------------------------------
__global__ void pack_bf16_k(const float* __restrict__ src, __bf16* __restrict__ dst,
                            int R, int C, long srs, long scs, int Rv, int Cv,
                            const float* __restrict__ rowscale, long sBatch, long dBatch)
{
  long b = blockIdx.z;
  long total = (long)R * C;
  for (long idx = blockIdx.x * (long)blockDim.x + threadIdx.x; idx < total;
       idx += (long)gridDim.x * blockDim.x) {
    int r = (int)(idx / C), c = (int)(idx % C);
    float v = 0.f;
    if (r < Rv && c < Cv) {
      v = src[b * sBatch + r * srs + c * scs];
      if (rowscale) v *= rowscale[r];
    }
    dst[b * dBatch + idx] = (__bf16)v;
  }
}

__global__ void prep_bn_k(const float* g1, const float* b1, const float* m1, const float* v1,
                          const float* mcb,
                          const float* g2, const float* b2, const float* m2, const float* v2,
                          const float* rcb,
                          float* s1, float* b1e, float* s2c, float* t2c)
{
  int c = blockIdx.x * blockDim.x + threadIdx.x;
  if (c < 512) {
    float s = g1[c] * rsqrtf(v1[c] + 1e-3f);
    s1[c]  = s;
    b1e[c] = (mcb[c] - m1[c]) * s + b1[c];
    float s2 = g2[c] * rsqrtf(v2[c] + 1e-3f);
    s2c[c] = s2;
    t2c[c] = (rcb[c] - m2[c]) * s2 + b2[c];
  }
}

__global__ void sim_stats_k(const float* __restrict__ sim, float* __restrict__ mu,
                            float* __restrict__ rv)
{
  __shared__ float r1[256], r2[256];
  int b = blockIdx.x, tid = threadIdx.x;
  const float* p = sim + (size_t)b * 262144;
  float s = 0.f, ss = 0.f;
  for (int i = tid; i < 262144; i += 256) { float x = p[i]; s += x; ss += x * x; }
  r1[tid] = s; r2[tid] = ss;
  __syncthreads();
  for (int st = 128; st > 0; st >>= 1) {
    if (tid < st) { r1[tid] += r1[tid + st]; r2[tid] += r2[tid + st]; }
    __syncthreads();
  }
  if (tid == 0) {
    float mean = r1[0] * (1.f / 262144.f);
    float var  = r2[0] * (1.f / 262144.f) - mean * mean;
    mu[b] = mean;
    rv[b] = rsqrtf(var + 1e-3f);
  }
}

__global__ void softmax_k(const float* __restrict__ sim, const float* __restrict__ mu,
                          const float* __restrict__ rv, const float* __restrict__ psig,
                          const float* __restrict__ psib, __bf16* __restrict__ P)
{
  __shared__ float red[256];
  int b = blockIdx.y, srow = blockIdx.x, tid = threadIdx.x;
  const float* row = sim + ((size_t)b * 512 + srow) * 512;
  float g  = psig[0] * rv[b];
  float off = psib[0] - mu[b] * g;
  float v0 = fmaf(row[tid], g, off);
  float v1 = fmaf(row[tid + 256], g, off);
  red[tid] = fmaxf(v0, v1);
  __syncthreads();
  for (int st = 128; st > 0; st >>= 1) {
    if (tid < st) red[tid] = fmaxf(red[tid], red[tid + st]);
    __syncthreads();
  }
  float mx = red[0];
  __syncthreads();
  float e0 = __expf(v0 - mx), e1 = __expf(v1 - mx);
  red[tid] = e0 + e1;
  __syncthreads();
  for (int st = 128; st > 0; st >>= 1) {
    if (tid < st) red[tid] += red[tid + st];
    __syncthreads();
  }
  float inv = 1.f / red[0];
  __bf16* pr = P + ((size_t)b * 512 + srow) * 512;
  pr[tid]       = (__bf16)(e0 * inv);
  pr[tid + 256] = (__bf16)(e1 * inv);
}

// ---------------------------------------------------------------------------
extern "C" void kernel_launch(void* const* d_in, const int* in_sizes, int n_in,
                              void* d_out, int out_size, void* d_ws, size_t ws_size,
                              hipStream_t stream) {
  (void)in_sizes; (void)n_in; (void)out_size; (void)ws_size;
  const float* dec   = (const float*)d_in[0];
  const float* trans = (const float*)d_in[1];
  const float* pe_w  = (const float*)d_in[2];
  const float* pe_b  = (const float*)d_in[3];
  const float* mc_w  = (const float*)d_in[4];
  const float* mc_b  = (const float*)d_in[5];
  const float* bn1g  = (const float*)d_in[6];
  const float* bn1b  = (const float*)d_in[7];
  const float* bn1m  = (const float*)d_in[8];
  const float* bn1v  = (const float*)d_in[9];
  const float* wq    = (const float*)d_in[10];
  const float* wk    = (const float*)d_in[11];
  const float* wv    = (const float*)d_in[12];
  const float* wo    = (const float*)d_in[13];
  const float* psig  = (const float*)d_in[14];
  const float* psib  = (const float*)d_in[15];
  const float* rc_w  = (const float*)d_in[16];
  const float* rc_b  = (const float*)d_in[17];
  const float* bn2g  = (const float*)d_in[18];
  const float* bn2b  = (const float*)d_in[19];
  const float* bn2m  = (const float*)d_in[20];
  const float* bn2v  = (const float*)d_in[21];
  float* out = (float*)d_out;

  // bump allocator on workspace (256B aligned)
  char* base = (char*)d_ws;
  size_t off = 0;
  auto alloc = [&](size_t bytes) -> void* {
    void* p = base + off;
    off += (bytes + 255) & ~(size_t)255;
    return p;
  };
  __bf16* pe_bf   = (__bf16*)alloc(512UL * 32768 * 2);
  __bf16* mcw     = (__bf16*)alloc(512UL * 512 * 2);
  __bf16* wqT     = (__bf16*)alloc(512UL * 512 * 2);
  __bf16* wk_b    = (__bf16*)alloc(768UL * 512 * 2);
  __bf16* wvT     = (__bf16*)alloc(512UL * 768 * 2);
  __bf16* woT     = (__bf16*)alloc(512UL * 512 * 2);
  __bf16* rcw     = (__bf16*)alloc(512UL * 512 * 2);
  __bf16* W2      = (__bf16*)alloc(512UL * 512 * 2);
  __bf16* trans_b = (__bf16*)alloc(8UL * 224 * 768 * 2);
  __bf16* transT  = (__bf16*)alloc(8UL * 768 * 224 * 2);
  __bf16* dl      = (__bf16*)alloc(8UL * 512 * 224 * 2);
  __bf16* qb      = (__bf16*)alloc(8UL * 512 * 224 * 2);
  __bf16* kb      = (__bf16*)alloc(8UL * 224 * 512 * 2);
  __bf16* vb      = (__bf16*)alloc(8UL * 512 * 224 * 2);
  __bf16* O1      = (__bf16*)alloc(8UL * 512 * 224 * 2);
  __bf16* Pb      = (__bf16*)alloc(8UL * 512 * 512 * 2);
  float*  simf    = (float*)alloc(8UL * 512 * 512 * 4);
  float*  z14     = (float*)alloc(8UL * 512 * 224 * 4);
  float*  s1      = (float*)alloc(512 * 4);
  float*  b1e     = (float*)alloc(512 * 4);
  float*  s2c     = (float*)alloc(512 * 4);
  float*  t2c     = (float*)alloc(512 * 4);
  float*  mu      = (float*)alloc(8 * 4);
  float*  rv      = (float*)alloc(8 * 4);

  auto pgrid = [](long n, int z) {
    long blk = (n + 255) / 256; if (blk > 4096) blk = 4096;
    return dim3((unsigned)blk, 1, (unsigned)z);
  };

  // ---- prep: BN folds + weight packs ----
  prep_bn_k<<<dim3(2), 256, 0, stream>>>(bn1g, bn1b, bn1m, bn1v, mc_b,
                                         bn2g, bn2b, bn2m, bn2v, rc_b,
                                         s1, b1e, s2c, t2c);
  pack_bf16_k<<<pgrid(512L*512, 1), 256, 0, stream>>>(mc_w, mcw, 512, 512, 512, 1, 512, 512, s1, 0, 0);
  pack_bf16_k<<<pgrid(512L*32768, 1), 256, 0, stream>>>(pe_w, pe_bf, 512, 32768, 32768, 1, 512, 32768, nullptr, 0, 0);
  pack_bf16_k<<<pgrid(512L*512, 1), 256, 0, stream>>>(wq, wqT, 512, 512, 1, 512, 512, 512, nullptr, 0, 0);
  pack_bf16_k<<<pgrid(768L*512, 1), 256, 0, stream>>>(wk, wk_b, 768, 512, 512, 1, 768, 512, nullptr, 0, 0);
  pack_bf16_k<<<pgrid(512L*768, 1), 256, 0, stream>>>(wv, wvT, 512, 768, 1, 512, 512, 768, nullptr, 0, 0);
  pack_bf16_k<<<pgrid(512L*512, 1), 256, 0, stream>>>(wo, woT, 512, 512, 1, 512, 512, 512, nullptr, 0, 0);
  pack_bf16_k<<<pgrid(512L*512, 1), 256, 0, stream>>>(rc_w, rcw, 512, 512, 512, 1, 512, 512, nullptr, 0, 0);
  pack_bf16_k<<<pgrid(224L*768, 8), 256, 0, stream>>>(trans, trans_b, 224, 768, 768, 1, 196, 768, nullptr, 196L*768, 224L*768);
  pack_bf16_k<<<pgrid(768L*224, 8), 256, 0, stream>>>(trans, transT, 768, 224, 1, 768, 768, 196, nullptr, 196L*768, 768L*224);

  // W2[c,s] = sum_o rc_w[c,o] * wo[s,o]   (folds reconstruct conv into wo)
  gemm_bf16_k<<<dim3(4, 8, 1), 256, 0, stream>>>(rcw, woT, W2, 512, 512, 512,
                                                 512, 512, 512, 0, 0, 0, 0, nullptr, nullptr);
  // patch embedding -> dl[b][d][224] bf16
  pe_gemm_k<<<dim3(4, 7, 8), 256, 0, stream>>>(pe_bf, dec, pe_b, dl);
  // q[b][s][n] = wqT @ dl
  gemm_bf16_k<<<dim3(4, 4, 8), 256, 0, stream>>>(wqT, dl, qb, 512, 224, 512,
                                                 512, 224, 224, 0, 512L*224, 512L*224, 0, nullptr, nullptr);
  // k[b][n][s] = trans @ wk
  gemm_bf16_k<<<dim3(2, 8, 8), 256, 0, stream>>>(trans_b, wk_b, kb, 224, 512, 768,
                                                 768, 512, 512, 224L*768, 0, 224L*512, 0, nullptr, nullptr);
  // v[b][s][n] = wvT @ transT
  gemm_bf16_k<<<dim3(4, 4, 8), 256, 0, stream>>>(wvT, transT, vb, 512, 224, 768,
                                                 768, 224, 224, 0, 768L*224, 512L*224, 0, nullptr, nullptr);
  // sim[b][s][t] = q @ k   (f32)
  gemm_bf16_k<<<dim3(4, 8, 8), 256, 0, stream>>>(qb, kb, simf, 512, 512, 224,
                                                 224, 512, 512, 512L*224, 224L*512, 512L*512, 2, nullptr, nullptr);
  // instance norm stats + softmax -> P bf16
  sim_stats_k<<<dim3(8), 256, 0, stream>>>(simf, mu, rv);
  softmax_k<<<dim3(512, 8), 256, 0, stream>>>(simf, mu, rv, psig, psib, Pb);
  // O1[b][s][n] = P @ v
  gemm_bf16_k<<<dim3(4, 4, 8), 256, 0, stream>>>(Pb, vb, O1, 512, 224, 512,
                                                 512, 224, 224, 512L*512, 512L*224, 512L*224, 0, nullptr, nullptr);
  // z14[b][c][n] = relu(bn2(W2 @ O1 + rc_b))   (f32, at 14x14 resolution)
  gemm_bf16_k<<<dim3(4, 4, 8), 256, 0, stream>>>(W2, O1, z14, 512, 224, 512,
                                                 512, 224, 224, 0, 512L*224, 512L*224, 3, s2c, t2c);
  // fused: mask GEMM + bn1 + relu + upsampled multiply -> out
  mask_final_k<<<dim3(4, 196, 8), 256, 0, stream>>>(mcw, dec, b1e, z14, out);
}
